// FlaxSelectiveAttention_43688407335380
// MI455X (gfx1250) — matrix operand order repaired and checked by reading, not compiled, not run on hardware
//
#include <hip/hip_runtime.h>
#include <hip/hip_bf16.h>

// MI455X / gfx1250: wave32, WMMA 16x16x32 bf16 -> f32.
// Pipeline:
//   1) hidden fp32 -> bf16
//   2) 8 weight matrices: fp32 [K][N] -> bf16 transposed [N][K] (LDS tile transpose)
//   3) fused gather + QKV projections + 4-way softmax attention (9 waves / block,
//      one wave per 16x64 WMMA GEMM tile; gather folded into A-row pointers)
//   4) out = attn @ Wo  (WMMA, fp32 accumulate -> fp32 d_out)
// Workspace: 32MB hidden_bf16 + 16MB weights + 32MB attn_bf16 = 80MB.

#define B_  4
#define S_  4096
#define H_  1024
#define NH_ 16
#define HD_ 64

typedef __bf16 bf16_t;
typedef __attribute__((ext_vector_type(16))) __bf16 v16bf;
typedef __attribute__((ext_vector_type(8)))  __bf16 v8bf;
typedef __attribute__((ext_vector_type(8)))  float  v8f;

union FragU {
    v16bf v;
    v8bf  h[2];
    int4  q[2];
};

// ---------------------------------------------------------------- convert hidden
__global__ __launch_bounds__(256) void k_cvt_hidden(const float* __restrict__ src,
                                                    bf16_t* __restrict__ dst, int n4) {
    int i = blockIdx.x * blockDim.x + threadIdx.x;
    if (i >= n4) return;
    const float4 f = ((const float4*)src)[i];
    union { bf16_t b[4]; unsigned long long u; } o;
    o.b[0] = (bf16_t)f.x; o.b[1] = (bf16_t)f.y;
    o.b[2] = (bf16_t)f.z; o.b[3] = (bf16_t)f.w;
    ((unsigned long long*)dst)[i] = o.u;
}

// ------------------------------------------------- transpose + convert weights
struct WPtrs { const float* p[8]; };

__global__ __launch_bounds__(256) void k_wt_transpose(WPtrs ws, bf16_t* __restrict__ dst_base) {
    __shared__ float tile[32][33];
    const float* src = ws.p[blockIdx.z];
    bf16_t* dst = dst_base + (size_t)blockIdx.z * (H_ * H_);
    const int tx = threadIdx.x, ty = threadIdx.y;
    const int n0 = blockIdx.x * 32;   // output-row block (N)
    const int k0 = blockIdx.y * 32;   // input-row block (K)
    #pragma unroll
    for (int i = ty; i < 32; i += 8)
        tile[i][tx] = src[(size_t)(k0 + i) * H_ + (n0 + tx)];
    __syncthreads();
    #pragma unroll
    for (int i = ty; i < 32; i += 8)
        dst[(size_t)(n0 + i) * H_ + (k0 + tx)] = (bf16_t)tile[tx][i];
}

// ------------------------------------------------ fused gather + QKV + attention
// grid: (B*S/16, NH). block: 288 threads = 9 waves.
// wave 0: q GEMM; waves 1..4: k_j GEMMs (gathered rows); waves 5..8: v_j GEMMs.
__global__ __launch_bounds__(288) void k_qkv_attn(const bf16_t* __restrict__ hidden,
                                                  const int*    __restrict__ conn,
                                                  const bf16_t* __restrict__ wt,
                                                  bf16_t*       __restrict__ attn_out) {
    __shared__ float q_s[16 * 64];
    __shared__ float k_s[4 * 16 * 64];
    __shared__ float v_s[4 * 16 * 64];
    __shared__ float logit_s[16 * 4];

    const int tid  = threadIdx.x;
    const int wave = tid >> 5;          // 0..8
    const int lane = tid & 31;
    const int half = lane >> 4;         // ISA A/B fragment half (lanes 0-15 vs 16-31)
    const int mrow = lane & 15;         // M (or N) within the 16x16 tile
    const int tile = blockIdx.x;        // 0..1023 over (b, s-tile)
    const int head = blockIdx.y;        // 0..15
    const int b    = tile >> 8;
    const int s0   = (tile & 255) << 4;

    // Select weight matrix and per-lane A-row pointer (gather for k/v waves).
    const bf16_t* wsel;
    const bf16_t* arow;
    bool valid = true;
    if (wave == 0) {
        wsel = wt; // q weights (transposed)
        arow = hidden + ((size_t)(b * S_) + s0 + mrow) * H_;
    } else {
        const int isv = (wave >= 5) ? 1 : 0;
        const int j   = isv ? (wave - 5) : (wave - 1);          // slot 0..3
        int widx = (j == 0) ? 1 : ((j == 3) ? 3 : 2);           // must / may / may / next
        widx += isv * 3;                                        // k: 1..3, v: 4..6
        wsel = wt + (size_t)widx * (H_ * H_);
        const int gidx = conn[(((size_t)(b * S_) + s0 + mrow) << 2) + j];
        valid = (gidx >= 0);
        arow  = hidden + ((size_t)(b * S_) + (valid ? gidx : 0)) * H_;
    }

    const v8f zero = {0.f, 0.f, 0.f, 0.f, 0.f, 0.f, 0.f, 0.f};
    v8f acc[4] = {zero, zero, zero, zero};

    const int c0 = head * HD_;          // 64-column head block
    for (int k0 = 0; k0 < H_; k0 += 32) {
        FragU a;
        if (valid) {
            // 16-bit A 16x32 layout: two contiguous 8-elem chunks per lane.
            a.h[0] = *(const v8bf*)(arow + k0 + half * 8);
            a.h[1] = *(const v8bf*)(arow + k0 + 16 + half * 8);
        } else {
            a.q[0] = make_int4(0, 0, 0, 0);
            a.q[1] = make_int4(0, 0, 0, 0);
        }
        __builtin_prefetch(arow + k0 + 32, 0, 1);   // global_prefetch_b8
        #pragma unroll
        for (int nt = 0; nt < 4; ++nt) {
            // B 32x16 layout from transposed weights: two 8-elem K chunks per lane (same order as A).
            const bf16_t* bp = wsel + (size_t)(c0 + nt * 16 + mrow) * H_ + k0 + half * 8;
            FragU bb;
            bb.h[0] = *(const v8bf*)(bp);
            bb.h[1] = *(const v8bf*)(bp + 16);
            acc[nt] = __builtin_amdgcn_wmma_f32_16x16x32_bf16(
                false, a.v, false, bb.v, (short)0, acc[nt], false, false);
        }
    }

    // Scatter accumulators to LDS (C/D layout: lane element r -> M = r + half*8, N = mrow).
    float* out_s = (wave == 0) ? q_s
                 : (wave < 5)  ? (k_s + (wave - 1) * 1024)
                               : (v_s + (wave - 5) * 1024);
    #pragma unroll
    for (int nt = 0; nt < 4; ++nt)
        #pragma unroll
        for (int r = 0; r < 8; ++r)
            out_s[(r + half * 8) * 64 + nt * 16 + mrow] = acc[nt][r];
    __syncthreads();

    // logits[row][j] = (q . k_j) * 1/sqrt(64)
    if (tid < 64) {
        const int row = tid >> 2, j = tid & 3;
        float s = 0.f;
        #pragma unroll 8
        for (int d = 0; d < 64; ++d)
            s += q_s[row * 64 + d] * k_s[j * 1024 + row * 64 + d];
        logit_s[row * 4 + j] = s * 0.125f;
    }
    __syncthreads();

    // softmax over 4 + weighted V, store bf16 attn rows for the output GEMM.
    for (int idx = tid; idx < 16 * 64; idx += 288) {
        const int row = idx >> 6, d = idx & 63;
        const float l0 = logit_s[row * 4 + 0], l1 = logit_s[row * 4 + 1];
        const float l2 = logit_s[row * 4 + 2], l3 = logit_s[row * 4 + 3];
        const float mx = fmaxf(fmaxf(l0, l1), fmaxf(l2, l3));
        const float e0 = __expf(l0 - mx), e1 = __expf(l1 - mx);
        const float e2 = __expf(l2 - mx), e3 = __expf(l3 - mx);
        const float inv = 1.f / (e0 + e1 + e2 + e3);
        const float o = (e0 * v_s[0 * 1024 + row * 64 + d] +
                         e1 * v_s[1 * 1024 + row * 64 + d] +
                         e2 * v_s[2 * 1024 + row * 64 + d] +
                         e3 * v_s[3 * 1024 + row * 64 + d]) * inv;
        attn_out[((size_t)(b * S_) + s0 + row) * H_ + c0 + d] = (bf16_t)o;
    }
}

// ------------------------------------------------------------ out = attn @ Wo
// grid: (B*S/16, 2). block: 256 threads = 8 waves, each wave a 16x64 tile.
__global__ __launch_bounds__(256) void k_out_gemm(const bf16_t* __restrict__ attn,
                                                  const bf16_t* __restrict__ wt_o,
                                                  float* __restrict__ out) {
    const int tid  = threadIdx.x;
    const int wave = tid >> 5, lane = tid & 31;
    const int half = lane >> 4, mrow = lane & 15;
    const size_t r0 = (size_t)blockIdx.x * 16;
    const int c0 = (blockIdx.y * 8 + wave) * 64;
    const bf16_t* arow = attn + (r0 + mrow) * H_;

    const v8f zero = {0.f, 0.f, 0.f, 0.f, 0.f, 0.f, 0.f, 0.f};
    v8f acc[4] = {zero, zero, zero, zero};

    for (int k0 = 0; k0 < H_; k0 += 32) {
        FragU a;
        a.h[0] = *(const v8bf*)(arow + k0 + half * 8);
        a.h[1] = *(const v8bf*)(arow + k0 + 16 + half * 8);
        #pragma unroll
        for (int nt = 0; nt < 4; ++nt) {
            const bf16_t* bp = wt_o + (size_t)(c0 + nt * 16 + mrow) * H_ + k0 + half * 8;
            FragU bb;
            bb.h[0] = *(const v8bf*)(bp);
            bb.h[1] = *(const v8bf*)(bp + 16);
            acc[nt] = __builtin_amdgcn_wmma_f32_16x16x32_bf16(
                false, a.v, false, bb.v, (short)0, acc[nt], false, false);
        }
    }
    #pragma unroll
    for (int nt = 0; nt < 4; ++nt)
        #pragma unroll
        for (int r = 0; r < 8; ++r)
            out[(r0 + r + half * 8) * H_ + c0 + nt * 16 + mrow] = acc[nt][r];
}

// ------------------------------------------------------------------- launcher
extern "C" void kernel_launch(void* const* d_in, const int* in_sizes, int n_in,
                              void* d_out, int out_size, void* d_ws, size_t ws_size,
                              hipStream_t stream) {
    const float* hidden_f = (const float*)d_in[0];
    const int*   conn     = (const int*)d_in[1];

    char* ws = (char*)d_ws;
    bf16_t* hidden_bf = (bf16_t*)ws;                                  // 32 MB
    bf16_t* wt        = (bf16_t*)(ws + (size_t)33554432);             // 16 MB (8 x 1024x1024)
    bf16_t* attn_bf   = (bf16_t*)(ws + (size_t)33554432 + 16777216);  // 32 MB

    // 1) hidden fp32 -> bf16
    const int n4 = B_ * S_ * H_ / 4;
    k_cvt_hidden<<<n4 / 256, 256, 0, stream>>>(hidden_f, hidden_bf, n4);

    // 2) weights fp32 [K][N] -> bf16 transposed [N][K]
    WPtrs wp;
    wp.p[0] = (const float*)d_in[2];  // q_w
    wp.p[1] = (const float*)d_in[3];  // k_w_must
    wp.p[2] = (const float*)d_in[4];  // k_w_may
    wp.p[3] = (const float*)d_in[5];  // k_w_next
    wp.p[4] = (const float*)d_in[6];  // v_w_must
    wp.p[5] = (const float*)d_in[7];  // v_w_may
    wp.p[6] = (const float*)d_in[8];  // v_w_next
    wp.p[7] = (const float*)d_in[9];  // o_w
    k_wt_transpose<<<dim3(32, 32, 8), dim3(32, 8), 0, stream>>>(wp, wt);

    // 3) fused gather + QKV + attention
    k_qkv_attn<<<dim3((B_ * S_) / 16, NH_), 288, 0, stream>>>(hidden_bf, conn, wt, attn_bf);

    // 4) out = attn @ Wo
    k_out_gemm<<<dim3((B_ * S_) / 16, 2), 256, 0, stream>>>(
        attn_bf, wt + (size_t)7 * H_ * H_, (float*)d_out);
}
